// GCNNet2_7035156431270
// MI455X (gfx1250) — matrix operand order repaired and checked by reading, not compiled, not run on hardware
//
#include <hip/hip_runtime.h>
#include <hip/hip_bf16.h>

// ---------------------------------------------------------------------------
// GCN forward on gfx1250 (MI455X): WMMA f16->f32 GEMMs with async global->LDS
// weight staging, L2-resident edge scatter with atomics, fused BN epilogues.
// ---------------------------------------------------------------------------

typedef __attribute__((ext_vector_type(16))) _Float16 v16h;
typedef __attribute__((ext_vector_type(8)))  float    v8f;

#define NN 100000
#define EE 500000
#define GG 100
#define DD 146
#define KP 160      // D padded to multiple of 32 (K) / 16 (N)
#define CC 10
#define NODES_PER_G (NN / GG)
#define MTILES (NN / 16)   // 6250 exactly

union V16H { v16h v; uint4 q[2]; };

// ------------------------- small utility kernels ---------------------------

__global__ void k_zero_f32(float* __restrict__ p, size_t n) {
    size_t i = (size_t)blockIdx.x * blockDim.x + threadIdx.x;
    if (i < n) p[i] = 0.0f;
}

__global__ void k_degrees(const int* __restrict__ src, const int* __restrict__ dst,
                          float* __restrict__ deg_out, float* __restrict__ deg_in) {
    int e = blockIdx.x * blockDim.x + threadIdx.x;
    if (e >= EE) return;
    atomicAdd(&deg_out[src[e]], 1.0f);
    atomicAdd(&deg_in[dst[e]], 1.0f);
}

__global__ void k_norms(const float* __restrict__ deg_out, const float* __restrict__ deg_in,
                        float* __restrict__ norm_out, float* __restrict__ norm_in) {
    int i = blockIdx.x * blockDim.x + threadIdx.x;
    if (i >= NN) return;
    norm_out[i] = rsqrtf(fmaxf(deg_out[i], 1.0f));
    norm_in[i]  = rsqrtf(fmaxf(deg_in[i], 1.0f));
}

// Convert [N,146] f32 (optionally row-scaled) into padded [N,160] f16.
__global__ void k_cvtX(const float* __restrict__ X, const float* __restrict__ rowscale,
                       _Float16* __restrict__ Xh, int nrows) {
    size_t i = (size_t)blockIdx.x * blockDim.x + threadIdx.x;
    size_t total = (size_t)nrows * KP;
    if (i >= total) return;
    int r = (int)(i / KP);
    int k = (int)(i % KP);
    float v = 0.0f;
    if (k < DD) {
        v = X[(size_t)r * DD + k];
        if (rowscale) v *= rowscale[r];
    }
    Xh[i] = (_Float16)v;
}

// Transpose+pad weight [146,146] -> WT[160,160] f16 with WT[n*160+k] = W[k,n].
__global__ void k_cvtW(const float* __restrict__ W, _Float16* __restrict__ WT) {
    int i = blockIdx.x * blockDim.x + threadIdx.x;
    if (i >= KP * KP) return;
    int n = i / KP;   // output column of the GEMM
    int k = i % KP;   // reduction index
    float v = (n < DD && k < DD) ? W[(size_t)k * DD + n] : 0.0f;
    WT[i] = (_Float16)v;
}

// ----------------------------- WMMA GEMM -----------------------------------
// Y[N,146] = Xh[N,160] @ W[160,160] (+ bias), optionally * snorm[row].
// W^T (51.2KB) is staged into LDS via async global->LDS copies (ASYNCcnt),
// then each wave owns a 16-row strip: 5 A-fragments loaded once from global,
// B-fragments read from LDS, 5 chained v_wmma_f32_16x16x32_f16 per N-tile.
__global__ __launch_bounds__(256)
void k_gemm_wmma(const _Float16* __restrict__ Xh, const _Float16* __restrict__ WT,
                 const float* __restrict__ bias, const float* __restrict__ snorm,
                 float* __restrict__ Y) {
    __shared__ __align__(16) _Float16 sWT[KP * KP];

    // LDS byte address of sWT: for LDS-aperture flat addresses the low 32 bits
    // ARE the wave-relative LDS offset (ISA 10.2). The ptrtoint also escapes
    // the array so the async-asm writes below are not optimized away.
    const unsigned lds_base = (unsigned)(unsigned long long)(const void*)sWT;

    // Async copy W^T to LDS: 3200 x 16B chunks across 256 threads (ASYNCcnt).
    {
        const int tid = threadIdx.x;
        for (int idx = tid; idx < (KP * KP * 2) / 16; idx += 256) {
            unsigned lds_off = lds_base + (unsigned)(idx * 16);
            unsigned long long gaddr =
                (unsigned long long)(const void*)((const char*)WT + (size_t)idx * 16);
            asm volatile("global_load_async_to_lds_b128 %0, %1, off"
                         :: "v"(lds_off), "v"(gaddr) : "memory");
        }
        asm volatile("s_wait_asynccnt 0x0" ::: "memory");
    }
    __syncthreads();

    const int wave = threadIdx.x >> 5;
    const int lane = threadIdx.x & 31;
    const int mtile = blockIdx.x * 8 + wave;
    if (mtile >= MTILES) return;            // wave-uniform: EXEC all-ones inside

    const int lh = lane >> 4;               // lane half (0/1)
    const int lr = lane & 15;               // A: row M; B/C: column N

    // A fragments: 16x32 f16 tile per K-step. Lane (lh,lr) holds row M=lr,
    // K = [k0+8*lh .. +8) and [k0+16+8*lh .. +8)  -> two 16B global loads each.
    const _Float16* xrow = Xh + (size_t)(mtile * 16 + lr) * KP;
    V16H A[5];
#pragma unroll
    for (int kk = 0; kk < 5; ++kk) {
        const int k0 = kk * 32;
        A[kk].q[0] = *(const uint4*)(xrow + k0 + lh * 8);
        A[kk].q[1] = *(const uint4*)(xrow + k0 + 16 + lh * 8);
    }

    // Hoisted per-row epilogue scales (no branching inside the tile loop).
    float sc[8];
    if (snorm) {
#pragma unroll
        for (int v = 0; v < 8; ++v) sc[v] = snorm[mtile * 16 + lh * 8 + v];
    } else {
#pragma unroll
        for (int v = 0; v < 8; ++v) sc[v] = 1.0f;
    }

    const int kh = lh * 8;                  // B: lanes 0-15 hold K 0-7,16-23; 16-31 hold 8-15,24-31
#pragma unroll 1
    for (int nt = 0; nt < 10; ++nt) {
        const _Float16* wrow = sWT + (nt * 16 + lr) * KP;   // row n of W^T (in LDS)
        // Fetch all 5 B fragments first (ds_load clause), then chain 5 WMMAs.
        V16H B[5];
#pragma unroll
        for (int kk = 0; kk < 5; ++kk) {
            B[kk].q[0] = *(const uint4*)(wrow + kk * 32 + kh);
            B[kk].q[1] = *(const uint4*)(wrow + kk * 32 + 16 + kh);
        }
        v8f C = {0.f, 0.f, 0.f, 0.f, 0.f, 0.f, 0.f, 0.f};
#pragma unroll
        for (int kk = 0; kk < 5; ++kk)
            C = __builtin_amdgcn_wmma_f32_16x16x32_f16(
                    false, A[kk].v, false, B[kk].v, (short)0, C, false, false);

        const int col = nt * 16 + lr;
        if (col < DD) {
            const float bv = bias[col];
            float* yp = Y + (size_t)(mtile * 16 + lh * 8) * DD + col;
#pragma unroll
            for (int v = 0; v < 8; ++v)                 // C/D layout: VGPR v -> M = 8*lh+v
                yp[(size_t)v * DD] = (C[v] + bv) * sc[v];
        }
    }
}

// ------------------------- edge gather / scatter ---------------------------
// agg[dst] += h[src] * norm_out[src]; one block per edge, contiguous D access
// (h fits in the 192MB L2, so these atomics resolve on-chip).
__global__ void k_scatter(const float* __restrict__ h, const int* __restrict__ src,
                          const int* __restrict__ dst, const float* __restrict__ norm_out,
                          float* __restrict__ agg) {
    const int e = blockIdx.x;
    const int s = src[e];
    const int t = dst[e];
    const float no = norm_out[s];
    const float* hs = h + (size_t)s * DD;
    float* at = agg + (size_t)t * DD;
    for (int d = threadIdx.x; d < DD; d += blockDim.x)
        atomicAdd(&at[d], hs[d] * no);
}

// ------------------------------ batch norm ---------------------------------

__global__ void k_colstats(const float* __restrict__ Y,
                           float* __restrict__ colsum, float* __restrict__ colsq) {
    const int d = threadIdx.x;
    if (d >= DD) return;
    float s = 0.0f, sq = 0.0f;
    for (int r = blockIdx.x; r < NN; r += gridDim.x) {
        const float v = Y[(size_t)r * DD + d];
        s += v;
        sq += v * v;
    }
    atomicAdd(&colsum[d], s);
    atomicAdd(&colsq[d], sq);
}

__global__ void k_bnparams(const float* __restrict__ colsum, const float* __restrict__ colsq,
                           float* __restrict__ mu, float* __restrict__ rs) {
    const int d = threadIdx.x;
    if (d >= DD) return;
    const float m = colsum[d] * (1.0f / NN);
    const float v = colsq[d] * (1.0f / NN) - m * m;
    mu[d] = m;
    rs[d] = rsqrtf(v + 1e-5f);
}

__global__ void k_finalize(const float* __restrict__ hin, const float* __restrict__ Y,
                           const float* __restrict__ gamma, const float* __restrict__ beta,
                           const float* __restrict__ mu, const float* __restrict__ rs,
                           float* __restrict__ hout) {
    const size_t i = (size_t)blockIdx.x * blockDim.x + threadIdx.x;
    if (i >= (size_t)NN * DD) return;
    const int d = (int)(i % DD);
    const float t = (Y[i] - mu[d]) * rs[d] * gamma[d] + beta[d];
    hout[i] = hin[i] + fmaxf(t, 0.0f);
}

// ------------------------------- readout -----------------------------------

__global__ void k_graphmean(const float* __restrict__ h, float* __restrict__ hg) {
    const int g = blockIdx.x;
    const int d = threadIdx.x;
    if (d >= DD) return;
    float s = 0.0f;
    const float* base = h + (size_t)g * NODES_PER_G * DD;
    for (int r = 0; r < NODES_PER_G; ++r) s += base[(size_t)r * DD + d];
    hg[(size_t)g * DD + d] = s * (1.0f / NODES_PER_G);
}

__global__ void k_mlp(const float* __restrict__ in, const float* __restrict__ W,
                      const float* __restrict__ b, float* __restrict__ out,
                      int IN, int OUT, int relu) {
    __shared__ float row[KP];
    const int g = blockIdx.x;
    for (int i = threadIdx.x; i < IN; i += blockDim.x) row[i] = in[(size_t)g * IN + i];
    __syncthreads();
    for (int o = threadIdx.x; o < OUT; o += blockDim.x) {
        float acc = b[o];
        for (int i = 0; i < IN; ++i) acc += row[i] * W[(size_t)i * OUT + o];
        if (relu) acc = fmaxf(acc, 0.0f);
        out[(size_t)g * OUT + o] = acc;
    }
}

// ------------------------------- launcher ----------------------------------

static inline size_t align256(size_t x) { return (x + 255) & ~(size_t)255; }

extern "C" void kernel_launch(void* const* d_in, const int* in_sizes, int n_in,
                              void* d_out, int out_size, void* d_ws, size_t ws_size,
                              hipStream_t stream) {
    (void)in_sizes; (void)n_in; (void)out_size; (void)ws_size;

    const float* nodes_feat = (const float*)d_in[0];
    const float* snorm      = (const float*)d_in[1];
    const float* W_emb      = (const float*)d_in[2];
    const float* b_emb      = (const float*)d_in[3];
    const float* Ws         = (const float*)d_in[4];   // [4,146,146]
    const float* bs         = (const float*)d_in[5];   // [4,146]
    const float* gammas     = (const float*)d_in[6];
    const float* betas      = (const float*)d_in[7];
    const float* W1         = (const float*)d_in[8];
    const float* b1         = (const float*)d_in[9];
    const float* W2         = (const float*)d_in[10];
    const float* b2         = (const float*)d_in[11];
    const float* W3         = (const float*)d_in[12];
    const float* b3         = (const float*)d_in[13];
    const int*   src        = (const int*)d_in[14];
    const int*   dst        = (const int*)d_in[15];
    float* out = (float*)d_out;

    // workspace carve-up
    char* ws = (char*)d_ws;
    size_t off = 0;
    const size_t NSf = (size_t)NN * DD * sizeof(float);
    float* P0 = (float*)(ws + off); off = align256(off + NSf);
    float* P1 = (float*)(ws + off); off = align256(off + NSf);
    float* Yb = (float*)(ws + off); off = align256(off + NSf);
    _Float16* Xh = (_Float16*)(ws + off); off = align256(off + (size_t)NN * KP * sizeof(_Float16));
    _Float16* WT = (_Float16*)(ws + off); off = align256(off + (size_t)KP * KP * sizeof(_Float16));
    float* deg_out = (float*)(ws + off); off = align256(off + NN * sizeof(float));
    float* deg_in  = (float*)(ws + off); off = align256(off + NN * sizeof(float));
    float* norm_out = (float*)(ws + off); off = align256(off + NN * sizeof(float));
    float* norm_in  = (float*)(ws + off); off = align256(off + NN * sizeof(float));
    float* colsum = (float*)(ws + off); off = align256(off + KP * sizeof(float));
    float* colsq  = (float*)(ws + off); off = align256(off + KP * sizeof(float));
    float* muv    = (float*)(ws + off); off = align256(off + KP * sizeof(float));
    float* rsv    = (float*)(ws + off); off = align256(off + KP * sizeof(float));
    float* hg = (float*)(ws + off); off = align256(off + (size_t)GG * DD * sizeof(float));
    float* x1 = (float*)(ws + off); off = align256(off + (size_t)GG * 73 * sizeof(float));
    float* x2 = (float*)(ws + off); off = align256(off + (size_t)GG * 36 * sizeof(float));

    const size_t ND = (size_t)NN * DD;
    const int gemm_blocks = (MTILES + 7) / 8;

    // degrees + norms
    k_zero_f32<<<(int)((2 * (size_t)NN + 255) / 256), 256, 0, stream>>>(deg_out, 2 * (size_t)NN);
    k_degrees<<<(EE + 255) / 256, 256, 0, stream>>>(src, dst, deg_out, deg_in);
    k_norms<<<(NN + 255) / 256, 256, 0, stream>>>(deg_out, deg_in, norm_out, norm_in);

    // embedding: P0 = nodes_feat @ W_emb + b_emb
    k_cvtW<<<(KP * KP + 255) / 256, 256, 0, stream>>>(W_emb, WT);
    k_cvtX<<<(int)(((size_t)NN * KP + 255) / 256), 256, 0, stream>>>(nodes_feat, nullptr, Xh, NN);
    k_gemm_wmma<<<gemm_blocks, 256, 0, stream>>>(Xh, WT, b_emb, nullptr, P0);

    float* hcur = P0;
    float* hnxt = P1;
    for (int l = 0; l < 4; ++l) {
        // agg (in hnxt) = scatter-add of row-scaled gathers
        k_zero_f32<<<(int)((ND + 255) / 256), 256, 0, stream>>>(hnxt, ND);
        k_scatter<<<EE, 160, 0, stream>>>(hcur, src, dst, norm_out, hnxt);
        // Y = (agg * norm_in) @ W_l + b_l, then * snorm
        k_cvtX<<<(int)(((size_t)NN * KP + 255) / 256), 256, 0, stream>>>(hnxt, norm_in, Xh, NN);
        k_cvtW<<<(KP * KP + 255) / 256, 256, 0, stream>>>(Ws + (size_t)l * DD * DD, WT);
        k_gemm_wmma<<<gemm_blocks, 256, 0, stream>>>(Xh, WT, bs + (size_t)l * DD, snorm, Yb);
        // batch-norm stats + fused BN/relu/residual into hnxt
        k_zero_f32<<<2, 256, 0, stream>>>(colsum, 2 * (size_t)KP);
        k_colstats<<<512, 160, 0, stream>>>(Yb, colsum, colsq);
        k_bnparams<<<1, 160, 0, stream>>>(colsum, colsq, muv, rsv);
        k_finalize<<<(int)((ND + 255) / 256), 256, 0, stream>>>(
            hcur, Yb, gammas + (size_t)l * DD, betas + (size_t)l * DD, muv, rsv, hnxt);
        float* t = hcur; hcur = hnxt; hnxt = t;
    }

    // readout
    k_graphmean<<<GG, 160, 0, stream>>>(hcur, hg);
    k_mlp<<<GG, 128, 0, stream>>>(hg, W1, b1, x1, DD, DD / 2, 1);
    k_mlp<<<GG, 128, 0, stream>>>(x1, W2, b2, x2, DD / 2, DD / 4, 1);
    k_mlp<<<GG, 64, 0, stream>>>(x2, W3, b3, out, DD / 4, CC, 0);
}
